// GraphSSM_11699490914487
// MI455X (gfx1250) — matrix operand order repaired and checked by reading, not compiled, not run on hardware
//
#include <hip/hip_runtime.h>

typedef __bf16 bf16_t;
typedef bf16_t v16bf __attribute__((ext_vector_type(16)));
typedef bf16_t v8bf  __attribute__((ext_vector_type(8)));
typedef bf16_t v4bf  __attribute__((ext_vector_type(4)));
typedef float  v8f   __attribute__((ext_vector_type(8)));

union V16 { v16bf v; v8bf h[2]; };

#define T_STEPS 32
#define DDIM    256
#define MBLK    64
#define PITCH   528                 // bytes per bf16 row: 264 elems (8-elem pad) -> 132 dwords, bank stride 4
#define SM_BW   0
#define SM_AW   (256 * PITCH)       // 135168
#define SM_X    (2 * 256 * PITCH)   // 270336
#define SMEM_BYTES (SM_X + MBLK * PITCH)  // 304128 <= 320KB

extern "C" __global__ __launch_bounds__(256)
void graphssm_kernel(const float* __restrict__ H,
                     const float* __restrict__ Aw,
                     const float* __restrict__ Bw,
                     const float* __restrict__ Bb,
                     float* __restrict__ out,
                     int N, int T)
{
    extern __shared__ char smem[];
    char* sBw = smem + SM_BW;
    char* sAw = smem + SM_AW;
    char* sX  = smem + SM_X;

    const int tid  = threadIdx.x;
    const int wave = tid >> 5;
    const int lane = tid & 31;
    const int hi   = lane >> 4;     // lane half (K-split selector)
    const int ln   = lane & 15;     // row/col within tile
    const int r2   = wave & 1;      // owns row tiles {2*r2, 2*r2+1}
    const int cq   = wave >> 1;     // owns col tiles {4*cq .. 4*cq+3}
    const int rowBase = blockIdx.x * MBLK;

    // ---- one-time: weights fp32 -> bf16 into LDS (row-major [n][k], pitch 528B)
#pragma unroll
    for (int m = 0; m < 2; ++m) {
        const float* W = m ? Aw : Bw;
        char* dst      = m ? sAw : sBw;
        for (int i = tid; i < 256 * 64; i += 256) {
            int row = i >> 6, c4 = i & 63;
            float4 w = *(const float4*)(W + row * 256 + c4 * 4);
            v4bf p = { (bf16_t)w.x, (bf16_t)w.y, (bf16_t)w.z, (bf16_t)w.w };
            *(v4bf*)(dst + row * PITCH + c4 * 8) = p;
        }
    }
    // ---- zero the H/S buffer (S_{-1} = 0)
    for (int i = tid * 16; i < MBLK * PITCH; i += 256 * 16)
        *(float4*)(sX + i) = float4{0.f, 0.f, 0.f, 0.f};

    // ---- per-lane bias (one value per owned col tile; same for both lane halves)
    float bias[4];
#pragma unroll
    for (int c = 0; c < 4; ++c) bias[c] = Bb[cq * 64 + c * 16 + ln];

    // ---- H prefetch registers: 16 x float4 per thread = 64 rows x 256 cols fp32
    float4 hv[16];
    auto loadH = [&](int t) {
#pragma unroll
        for (int i = 0; i < 16; ++i) {
            int f = tid + i * 256;
            int row = f >> 6, c4 = f & 63;
            int g = rowBase + row;
            hv[i] = (g < N)
                ? *(const float4*)(H + ((size_t)t * N + g) * 256 + c4 * 4)
                : float4{0.f, 0.f, 0.f, 0.f};
        }
    };

    // ---- bf16 WMMA GEMM over K=256: acc += X(64x256) @ W^T tile-wise
    auto gemm = [&](const char* xbuf, const char* wbuf, v8f (&acc)[2][4]) {
#pragma unroll
        for (int kb = 0; kb < 8; ++kb) {
            const int kOff = kb * 64;   // 32 K-values * 2 bytes
            V16 a[2];
#pragma unroll
            for (int i = 0; i < 2; ++i) {
                const char* ap = xbuf + ((r2 * 2 + i) * 16 + ln) * PITCH + kOff + hi * 16;
                a[i].h[0] = *(const v8bf*)(ap);
                a[i].h[1] = *(const v8bf*)(ap + 32);
            }
#pragma unroll
            for (int c = 0; c < 4; ++c) {
                const char* bp = wbuf + (cq * 64 + c * 16 + ln) * PITCH + kOff + hi * 16;
                V16 b;
                b.h[0] = *(const v8bf*)(bp);
                b.h[1] = *(const v8bf*)(bp + 32);
#pragma unroll
                for (int i = 0; i < 2; ++i)
                    acc[i][c] = __builtin_amdgcn_wmma_f32_16x16x32_bf16(
                        false, a[i].v, false, b.v, (short)0, acc[i][c], false, false);
            }
        }
    };

    loadH(0);
    __syncthreads();   // weights + zeroed S visible

    for (int t = 0; t < T; ++t) {
        v8f acc[2][4];
#pragma unroll
        for (int i = 0; i < 2; ++i)
#pragma unroll
            for (int c = 0; c < 4; ++c)
#pragma unroll
                for (int d = 0; d < 8; ++d) acc[i][c][d] = bias[c];

        // phase A: S_{t-1} @ A_w^T   (xbuf holds S_{t-1} bf16)
        gemm(sX, sAw, acc);
        __syncthreads();

        // commit prefetched H_t (fp32 regs -> bf16 LDS), then prefetch H_{t+1}
#pragma unroll
        for (int i = 0; i < 16; ++i) {
            int f = tid + i * 256;
            int row = f >> 6, c4 = f & 63;
            float4 w = hv[i];
            v4bf p = { (bf16_t)w.x, (bf16_t)w.y, (bf16_t)w.z, (bf16_t)w.w };
            *(v4bf*)(sX + row * PITCH + c4 * 8) = p;
        }
        if (t + 1 < T) loadH(t + 1);
        __syncthreads();

        // phase B: + H_t @ B_w^T  -> acc now holds S_t
        gemm(sX, sBw, acc);

        // store S_t (fp32) to global: C/D layout -> (row = tile*16 + d + 8*hi, col = tile*16 + ln)
#pragma unroll
        for (int i = 0; i < 2; ++i)
#pragma unroll
            for (int c = 0; c < 4; ++c)
#pragma unroll
                for (int d = 0; d < 8; ++d) {
                    int row = (r2 * 2 + i) * 16 + d + hi * 8;
                    int col = cq * 64 + c * 16 + ln;
                    int g = rowBase + row;
                    if (g < N)
                        out[((size_t)t * N + g) * 256 + col] = acc[i][c][d];
                }
        __syncthreads();   // all reads of H_t in xbuf complete

        // write S_t back to xbuf as bf16 for the next step's phase A
#pragma unroll
        for (int i = 0; i < 2; ++i)
#pragma unroll
            for (int c = 0; c < 4; ++c)
#pragma unroll
                for (int d = 0; d < 8; ++d) {
                    int row = (r2 * 2 + i) * 16 + d + hi * 8;
                    int col = cq * 64 + c * 16 + ln;
                    *(bf16_t*)(sX + row * PITCH + col * 2) = (bf16_t)acc[i][c][d];
                }
        __syncthreads();
    }
}

extern "C" void kernel_launch(void* const* d_in, const int* in_sizes, int n_in,
                              void* d_out, int out_size, void* d_ws, size_t ws_size,
                              hipStream_t stream) {
    (void)n_in; (void)out_size; (void)d_ws; (void)ws_size;
    const float* H  = (const float*)d_in[0];
    const float* Aw = (const float*)d_in[1];
    const float* Bw = (const float*)d_in[2];
    const float* Bb = (const float*)d_in[3];
    float* out = (float*)d_out;

    const int T = T_STEPS;
    const int N = in_sizes[0] / (T * DDIM);   // 10000

    hipFuncSetAttribute((const void*)graphssm_kernel,
                        hipFuncAttributeMaxDynamicSharedMemorySize, SMEM_BYTES);

    dim3 grid((N + MBLK - 1) / MBLK), block(256);
    hipLaunchKernelGGL(graphssm_kernel, grid, block, SMEM_BYTES, stream,
                       H, Aw, Bw, Bb, out, N, T);
}
